// GPTLanguageModel_66391604462096
// MI455X (gfx1250) — matrix operand order repaired and checked by reading, not compiled, not run on hardware
//
#include <hip/hip_runtime.h>
#include <stdint.h>

// Model constants (match reference)
#define Bn   4
#define Tn   1024
#define Vn   32000
#define En   1024
#define Hn   16
#define HSn  8
#define Ln   8
#define BTn  (Bn*Tn)          // 4096 rows
#define FEn  (4*En)           // 4096 ffn dim
#define LN_EPS 1e-5f

typedef __attribute__((ext_vector_type(16))) _Float16     v16h;
typedef __attribute__((ext_vector_type(8)))  float        v8f;
typedef __attribute__((ext_vector_type(4)))  unsigned int v4u;

// ---------------------------------------------------------------------------
// CDNA5 async global->LDS (ASYNCcnt) helpers. LDS dest offset comes from a
// per-lane VGPR (byte offset into workgroup LDS); GV addressing (saddr=off).
// ---------------------------------------------------------------------------
__device__ __forceinline__ void async_g2lds_b128(unsigned ldsoff, const void* gptr) {
  unsigned long long ga = (unsigned long long)gptr;
  asm volatile("global_load_async_to_lds_b128 %0, %1, off"
               :: "v"(ldsoff), "v"(ga) : "memory");
}
__device__ __forceinline__ void wait_async0() {
  asm volatile("s_wait_asynccnt 0x0" ::: "memory");
}

// ---------------------------------------------------------------------------
// f32 -> f16 convert (grid-stride), plain layout copy
// ---------------------------------------------------------------------------
__global__ void k_cvt_f16(const float* __restrict__ in, _Float16* __restrict__ out,
                          long long n) {
  long long i = (long long)blockIdx.x * blockDim.x + threadIdx.x;
  long long st = (long long)gridDim.x * blockDim.x;
  for (; i < n; i += st) out[i] = (_Float16)in[i];
}

// f32 (R x C row-major) -> f16 transposed (C x R row-major)
__global__ void k_cvt_t(const float* __restrict__ in, _Float16* __restrict__ out,
                        int R, int C) {
  long long n = (long long)R * C;
  long long i = (long long)blockIdx.x * blockDim.x + threadIdx.x;
  long long st = (long long)gridDim.x * blockDim.x;
  for (; i < n; i += st) {
    int r = (int)(i / C), c = (int)(i - (long long)r * C);
    out[(size_t)c * R + r] = (_Float16)in[i];
  }
}

// Wq/Wk/Wv layer slice: (H, E, HS) -> transposed weight (H*HS, E) f16
__global__ void k_cvt_qkv_t(const float* __restrict__ in, _Float16* __restrict__ out) {
  int n = blockIdx.x * blockDim.x + threadIdx.x;      // n < H*E*HS = 131072
  int h = n / (En * HSn);
  int r = n - h * (En * HSn);
  int e = r / HSn;
  int d = r - e * HSn;
  out[(size_t)(h * HSn + d) * En + e] = (_Float16)in[n];
}

// ---------------------------------------------------------------------------
// Embedding: x[b,t,:] = tok_emb[idx[b,t],:] + pos_emb[t,:]
// ---------------------------------------------------------------------------
__global__ void k_embed(const int* __restrict__ idx, const float* __restrict__ tok,
                        const float* __restrict__ pos, float* __restrict__ x) {
  long long i = (long long)blockIdx.x * blockDim.x + threadIdx.x;
  long long st = (long long)gridDim.x * blockDim.x;
  const long long n = (long long)BTn * En;
  for (; i < n; i += st) {
    int e  = (int)(i & (En - 1));
    int bt = (int)(i >> 10);
    int t  = bt & (Tn - 1);
    x[i] = tok[(size_t)idx[bt] * En + e] + pos[(size_t)t * En + e];
  }
}

// ---------------------------------------------------------------------------
// LayerNorm: (BTn, En) f32 -> f16, one block (256 thr) per row
// ---------------------------------------------------------------------------
__global__ void k_ln(const float* __restrict__ X, const float* __restrict__ sc,
                     const float* __restrict__ bi, _Float16* __restrict__ O) {
  const int row = blockIdx.x;
  const float* xp = X + (size_t)row * En;
  float s = 0.f, q = 0.f;
  for (int i = threadIdx.x; i < En; i += 256) { float v = xp[i]; s += v; q += v * v; }
  __shared__ float r1[256], r2[256];
  r1[threadIdx.x] = s; r2[threadIdx.x] = q; __syncthreads();
  for (int st = 128; st > 0; st >>= 1) {
    if ((int)threadIdx.x < st) { r1[threadIdx.x] += r1[threadIdx.x + st];
                                 r2[threadIdx.x] += r2[threadIdx.x + st]; }
    __syncthreads();
  }
  const float mu  = r1[0] * (1.f / En);
  const float var = r2[0] * (1.f / En) - mu * mu;
  const float rs  = rsqrtf(var + LN_EPS);
  for (int i = threadIdx.x; i < En; i += 256)
    O[(size_t)row * En + i] = (_Float16)((xp[i] - mu) * rs * sc[i] + bi[i]);
}

// ---------------------------------------------------------------------------
// WMMA f16 GEMM: C[M,N] = A[M,K] @ B^T(+bias)(+relu)(+resid), f32 accumulate.
// A: M x K row-major f16 (activations), per-wave global_load_b128, register-
//    pipelined one K-step ahead (two explicit Frag vars, K-loop unrolled x2
//    so all fragment/buffer selection is static -> no alloca promotion).
// B: N x K row-major f16 (weights pre-transposed). The 128-col B strip K-slice
//    (128x32 f16 = 8KB) is staged to LDS once per block per K-step via
//    global_load_async_to_lds_b128 (double-buffered, ASYNCcnt + barrier).
// Block = 256 thr (8 waves) = 128(M) x 128(N) C tile; each wave computes a
// 16x128 strip: one A fragment feeds 8 v_wmma. B fragments ping-pong through
// two registers: ds_load(tt+1) issued before wmma(tt).
// LDS column stride padded to 40 halves (80B = 20 banks) -> conflict-free
// 16B lane reads. Fragment layouts per CDNA5 ISA 7.12.2.
// Requires K % 64 == 0 (true for all call sites: 128, 1024, 4096).
// ---------------------------------------------------------------------------
#define BNt       128                     // block N tile
#define LB_STRIDE 40                      // halves per column (32 + 8 pad)
#define LB_BUF    (BNt * LB_STRIDE)       // halves per buffer (5120)

template <bool BIAS, bool RELU, bool RESID, bool F16OUT>
__global__ void k_gemm(const _Float16* __restrict__ A, const _Float16* __restrict__ B,
                       float* __restrict__ Cf, _Float16* __restrict__ Ch,
                       const float* __restrict__ bias,
                       const float* __restrict__ resid,
                       int M, int N, int K) {
  __shared__ _Float16 lb[2 * LB_BUF];     // only LDS object -> offset 0 (20.5KB)

  const int tid  = threadIdx.x;
  const int lane = tid & 31, wave = tid >> 5;
  const int tilesN = N >> 7;
  const int bm = blockIdx.x / tilesN, bn = blockIdx.x % tilesN;
  const int n0 = bn * BNt;
  const int g  = lane >> 4;               // half-wave group
  const int lm = lane & 15;
  const int rowA = bm * 128 + wave * 16 + lm;

  // staging map: 512 units of 16B (128 cols x 4 segs); 2 units per thread
  const int c0 = tid >> 2,          s0 = tid & 3;
  const int c1 = (tid + 256) >> 2,  s1 = tid & 3;    // +256 units: col + 64
  const _Float16* sg0 = B + (size_t)(n0 + c0) * K + s0 * 8;
  const _Float16* sg1 = B + (size_t)(n0 + c1) * K + s1 * 8;
  const unsigned  sl0 = (unsigned)((c0 * LB_STRIDE + s0 * 8) * 2);
  const unsigned  sl1 = (unsigned)((c1 * LB_STRIDE + s1 * 8) * 2);

  union Frag { v16h h; v4u q[2]; };
  Frag a0, a1, b0, b1;
  v8f c[8] = {{}, {}, {}, {}, {}, {}, {}, {}};
  const _Float16* Ap = A + (size_t)rowA * K;

  // 8-tile wmma chain from one LDS buffer; B frags ping-pong b0/b1 (static)
  auto compute = [&](const _Float16* lbase, Frag& af) {
    {
      const _Float16* lp = lbase + lm * LB_STRIDE + g * 8;
      b0.q[0] = *(const v4u*)(lp);
      b0.q[1] = *(const v4u*)(lp + 16);
    }
#pragma unroll
    for (int tt = 0; tt < 8; ++tt) {
      Frag& bc = (tt & 1) ? b1 : b0;
      Frag& bnx = (tt & 1) ? b0 : b1;
      if (tt < 7) {
        const _Float16* lp = lbase + ((tt + 1) * 16 + lm) * LB_STRIDE + g * 8;
        bnx.q[0] = *(const v4u*)(lp);
        bnx.q[1] = *(const v4u*)(lp + 16);
      }
      c[tt] = __builtin_amdgcn_wmma_f32_16x16x32_f16(false, af.h, false, bc.h,
                                                     (short)0, c[tt], false, false);
    }
  };

  // prologue: stage B K-slice 0 into buffer 0; load A fragment for kk=0
  async_g2lds_b128(sl0, sg0);
  async_g2lds_b128(sl1, sg1);
  a0.q[0] = *(const v4u*)(Ap + g * 8);
  a0.q[1] = *(const v4u*)(Ap + 16 + g * 8);

  for (int kk = 0; kk < K; kk += 64) {
    // ---- phase 0: consume buffer 0 with a0; prefetch kk+32 ----
    wait_async0();
    __syncthreads();
    async_g2lds_b128(sl0 + LB_BUF * 2, sg0 + kk + 32);   // kk+32 < K always
    async_g2lds_b128(sl1 + LB_BUF * 2, sg1 + kk + 32);
    a1.q[0] = *(const v4u*)(Ap + kk + 32 + g * 8);
    a1.q[1] = *(const v4u*)(Ap + kk + 48 + g * 8);
    compute(lb, a0);

    // ---- phase 1: consume buffer 1 with a1; prefetch kk+64 (clamped) ----
    wait_async0();
    __syncthreads();
    const int knext = (kk + 64 < K) ? kk + 64 : 0;       // dummy on last iter
    async_g2lds_b128(sl0, sg0 + knext);
    async_g2lds_b128(sl1, sg1 + knext);
    a0.q[0] = *(const v4u*)(Ap + knext + g * 8);
    a0.q[1] = *(const v4u*)(Ap + knext + 16 + g * 8);
    compute(lb + LB_BUF, a1);
  }

#pragma unroll
  for (int tt = 0; tt < 8; ++tt) {
    const int col = n0 + tt * 16 + lm;
    float bcol = 0.f;
    if (BIAS) bcol = bias[col];
#pragma unroll
    for (int r = 0; r < 8; ++r) {
      int row = bm * 128 + wave * 16 + g * 8 + r;   // C VGPR r -> row g*8+r
      size_t o = (size_t)row * N + col;
      float v = c[tt][r];
      if (BIAS)  v += bcol;
      if (RELU)  v = v > 0.f ? v : 0.f;
      if (RESID) v += resid[o];
      if (F16OUT) Ch[o] = (_Float16)v;
      else        Cf[o] = v;
    }
  }
}

// ---------------------------------------------------------------------------
// Causal attention, online softmax. One wave32 per (b,h,t) row; HS=8.
// Q/K/V are (B*T, H*HS) f32; output att (B*T, H*HS) f16.
// ---------------------------------------------------------------------------
__global__ void k_attn(const float* __restrict__ Q, const float* __restrict__ K,
                       const float* __restrict__ V, _Float16* __restrict__ O) {
  const int wave = threadIdx.x >> 5, lane = threadIdx.x & 31;
  const int row = blockIdx.x * 8 + wave;            // row in [0, B*H*T)
  const int t  = row & (Tn - 1);
  const int bh = row >> 10;
  const int b  = bh >> 4;
  const int h  = bh & 15;
  const float scale = 0.3535533906f;                // 8^-0.5

  float q[HSn];
  const float* qp = Q + ((size_t)(b * Tn + t) * (Hn * HSn) + h * HSn);
#pragma unroll
  for (int d = 0; d < HSn; ++d) q[d] = qp[d];

  float m = -1e30f, l = 0.f, acc[HSn] = {};
  for (int s = lane; s <= t; s += 32) {
    const float* kp = K + ((size_t)(b * Tn + s) * (Hn * HSn) + h * HSn);
    float dot = 0.f;
#pragma unroll
    for (int d = 0; d < HSn; ++d) dot += q[d] * kp[d];
    dot *= scale;
    float mn = fmaxf(m, dot);
    float em = __expf(m - mn), ed = __expf(dot - mn);
    l = l * em + ed;
    const float* vp = V + ((size_t)(b * Tn + s) * (Hn * HSn) + h * HSn);
#pragma unroll
    for (int d = 0; d < HSn; ++d) acc[d] = acc[d] * em + ed * vp[d];
    m = mn;
  }
  // cross-lane merge of (m, l, acc)
#pragma unroll
  for (int off = 16; off > 0; off >>= 1) {
    float m2 = __shfl_xor(m, off, 32);
    float l2 = __shfl_xor(l, off, 32);
    float mn = fmaxf(m, m2);
    float e1 = __expf(m - mn), e2 = __expf(m2 - mn);
    l = l * e1 + l2 * e2;
#pragma unroll
    for (int d = 0; d < HSn; ++d) {
      float a2 = __shfl_xor(acc[d], off, 32);
      acc[d] = acc[d] * e1 + a2 * e2;
    }
    m = mn;
  }
  if (lane == 0) {
    float inv = 1.f / l;
    _Float16* op = O + ((size_t)(b * Tn + t) * (Hn * HSn) + h * HSn);
#pragma unroll
    for (int d = 0; d < HSn; ++d) op[d] = (_Float16)(acc[d] * inv);
  }
}

// ---------------------------------------------------------------------------
// Loss: -mean over rows of log_softmax(logits)[target]
// ---------------------------------------------------------------------------
__global__ void k_zero(float* p) { p[0] = 0.f; }

__global__ void k_loss(const float* __restrict__ logits, const int* __restrict__ tgt,
                       float* loss) {
  const int row = blockIdx.x;                       // BTn rows
  const float* lp = logits + (size_t)row * Vn;
  __shared__ float red[256];
  float mx = -1e30f;
  for (int i = threadIdx.x; i < Vn; i += 256) mx = fmaxf(mx, lp[i]);
  red[threadIdx.x] = mx; __syncthreads();
  for (int st = 128; st > 0; st >>= 1) {
    if ((int)threadIdx.x < st) red[threadIdx.x] = fmaxf(red[threadIdx.x], red[threadIdx.x + st]);
    __syncthreads();
  }
  mx = red[0]; __syncthreads();
  float sum = 0.f;
  for (int i = threadIdx.x; i < Vn; i += 256) sum += __expf(lp[i] - mx);
  red[threadIdx.x] = sum; __syncthreads();
  for (int st = 128; st > 0; st >>= 1) {
    if ((int)threadIdx.x < st) red[threadIdx.x] += red[threadIdx.x + st];
    __syncthreads();
  }
  if (threadIdx.x == 0) {
    float lse = mx + logf(red[0]);
    float li  = lse - lp[tgt[row]];
    atomicAdd(loss, li * (1.0f / BTn));
  }
}

// ---------------------------------------------------------------------------
// Host launch
// ---------------------------------------------------------------------------
static inline int gemm_blocks(int M, int N) { return (M / 128) * (N / 128); }

extern "C" void kernel_launch(void* const* d_in, const int* in_sizes, int n_in,
                              void* d_out, int out_size, void* d_ws, size_t ws_size,
                              hipStream_t stream) {
  const int*   idx     = (const int*)  d_in[0];
  const int*   targets = (const int*)  d_in[1];
  const float* tok_emb = (const float*)d_in[2];
  const float* pos_emb = (const float*)d_in[3];
  const float* Wq      = (const float*)d_in[4];
  const float* Wk      = (const float*)d_in[5];
  const float* Wv      = (const float*)d_in[6];
  const float* Wo      = (const float*)d_in[7];
  const float* bo      = (const float*)d_in[8];
  const float* W1      = (const float*)d_in[9];
  const float* b1      = (const float*)d_in[10];
  const float* W2      = (const float*)d_in[11];
  const float* b2      = (const float*)d_in[12];
  const float* ln1_s   = (const float*)d_in[13];
  const float* ln1_b   = (const float*)d_in[14];
  const float* ln2_s   = (const float*)d_in[15];
  const float* ln2_b   = (const float*)d_in[16];
  const float* lnf_s   = (const float*)d_in[17];
  const float* lnf_b   = (const float*)d_in[18];

  float* logits = (float*)d_out;                       // BTn * Vn
  float* loss   = logits + (size_t)BTn * Vn;           // 1 float

  // workspace carve
  uintptr_t p = (uintptr_t)d_ws;
  auto take = [&](size_t bytes) -> void* {
    void* r = (void*)p;
    p = (p + bytes + 255) & ~(uintptr_t)255;
    return r;
  };
  float*    x     = (float*)    take((size_t)BTn * En * 4);
  _Float16* h16   = (_Float16*) take((size_t)BTn * En * 2);
  float*    q     = (float*)    take((size_t)BTn * Hn * HSn * 4);
  float*    kbuf  = (float*)    take((size_t)BTn * Hn * HSn * 4);
  float*    vbuf  = (float*)    take((size_t)BTn * Hn * HSn * 4);
  _Float16* att16 = (_Float16*) take((size_t)BTn * Hn * HSn * 2);
  _Float16* mlp16 = (_Float16*) take((size_t)BTn * FEn * 2);
  _Float16* wq16  = (_Float16*) take((size_t)En * Hn * HSn * 2);  // (128, E) N x K
  _Float16* wk16  = (_Float16*) take((size_t)En * Hn * HSn * 2);
  _Float16* wv16  = (_Float16*) take((size_t)En * Hn * HSn * 2);
  _Float16* wo16  = (_Float16*) take((size_t)Hn * HSn * En * 2);  // (E, 128)
  _Float16* w116  = (_Float16*) take((size_t)En * FEn * 2);       // (4E, E)
  _Float16* w216  = (_Float16*) take((size_t)FEn * En * 2);       // (E, 4E)
  _Float16* te16  = (_Float16*) take((size_t)Vn * En * 2);        // (V, E) natural N x K

  // tok_emb -> f16 (weight-tied head; already N x K layout)
  k_cvt_f16<<<4096, 256, 0, stream>>>(tok_emb, te16, (long long)Vn * En);
  // embeddings
  k_embed<<<4096, 256, 0, stream>>>(idx, tok_emb, pos_emb, x);

  for (int l = 0; l < Ln; ++l) {
    const size_t wqkv_off = (size_t)l * Hn * En * HSn;   // 131072
    k_cvt_qkv_t<<<512, 256, 0, stream>>>(Wq + wqkv_off, wq16);
    k_cvt_qkv_t<<<512, 256, 0, stream>>>(Wk + wqkv_off, wk16);
    k_cvt_qkv_t<<<512, 256, 0, stream>>>(Wv + wqkv_off, wv16);
    k_cvt_t<<<512, 256, 0, stream>>>(Wo + (size_t)l * Hn * HSn * En, wo16,
                                     Hn * HSn, En);          // (128,E) -> (E,128)
    k_cvt_t<<<4096, 256, 0, stream>>>(W1 + (size_t)l * En * FEn, w116,
                                      En, FEn);              // (E,4E) -> (4E,E)
    k_cvt_t<<<4096, 256, 0, stream>>>(W2 + (size_t)l * FEn * En, w216,
                                      FEn, En);              // (4E,E) -> (E,4E)

    // ln1 -> h16
    k_ln<<<BTn, 256, 0, stream>>>(x, ln1_s + (size_t)l * En, ln1_b + (size_t)l * En, h16);

    // q,k,v = h16 @ W{q,k,v}  (M=4096, N=128, K=1024)
    k_gemm<false, false, false, false><<<gemm_blocks(BTn, Hn * HSn), 256, 0, stream>>>(
        h16, wq16, q, nullptr, nullptr, nullptr, BTn, Hn * HSn, En);
    k_gemm<false, false, false, false><<<gemm_blocks(BTn, Hn * HSn), 256, 0, stream>>>(
        h16, wk16, kbuf, nullptr, nullptr, nullptr, BTn, Hn * HSn, En);
    k_gemm<false, false, false, false><<<gemm_blocks(BTn, Hn * HSn), 256, 0, stream>>>(
        h16, wv16, vbuf, nullptr, nullptr, nullptr, BTn, Hn * HSn, En);

    // attention (B*H*T rows, 8 waves/block)
    k_attn<<<(Bn * Hn * Tn) / 8, 256, 0, stream>>>(q, kbuf, vbuf, att16);

    // x += att @ Wo + bo  (M=4096, N=1024, K=128)
    k_gemm<true, false, true, false><<<gemm_blocks(BTn, En), 256, 0, stream>>>(
        att16, wo16, x, nullptr, bo + (size_t)l * En, x, BTn, En, Hn * HSn);

    // ln2 -> h16
    k_ln<<<BTn, 256, 0, stream>>>(x, ln2_s + (size_t)l * En, ln2_b + (size_t)l * En, h16);

    // mlp16 = relu(h16 @ W1 + b1)  (M=4096, N=4096, K=1024)
    k_gemm<true, true, false, true><<<gemm_blocks(BTn, FEn), 256, 0, stream>>>(
        h16, w116, nullptr, mlp16, b1 + (size_t)l * FEn, nullptr, BTn, FEn, En);

    // x += mlp16 @ W2 + b2  (M=4096, N=1024, K=4096)
    k_gemm<true, false, true, false><<<gemm_blocks(BTn, En), 256, 0, stream>>>(
        mlp16, w216, x, nullptr, b2 + (size_t)l * En, x, BTn, En, FEn);
  }

  // final LN -> h16
  k_ln<<<BTn, 256, 0, stream>>>(x, lnf_s, lnf_b, h16);

  // logits = h16 @ tok_emb^T  (M=4096, N=32000, K=1024)
  k_gemm<false, false, false, false><<<gemm_blocks(BTn, Vn), 256, 0, stream>>>(
      h16, te16, logits, nullptr, nullptr, nullptr, BTn, Vn, En);

  // loss
  k_zero<<<1, 1, 0, stream>>>(loss);
  k_loss<<<BTn, 256, 0, stream>>>(logits, targets, loss);
}
